// EdgeFeature_41549513621914
// MI455X (gfx1250) — matrix operand order repaired and checked by reading, not compiled, not run on hardware
//
#include <hip/hip_runtime.h>

typedef __attribute__((ext_vector_type(16))) _Float16 v16h;
typedef __attribute__((ext_vector_type(8)))  float    v8f;

union V16H { v16h v; _Float16 e[16]; };
union V8F  { v8f  v; float    e[8];  };

#define NPTS 4096
#define DIM  64
#define KNN  20
#define TILE 16

__global__ __launch_bounds__(256, 1)
void edge_feature_knn_kernel(const float* __restrict__ xg, float* __restrict__ out) {
    // Dynamic LDS: dist[16][4096] f32 | Xr[16][64] f32 | rowsq[16] f32 | nn[16][20] i32
    extern __shared__ float smem[];
    float* dist  = smem;                    // 16*4096 f32 = 256 KB
    float* Xr    = dist + TILE * NPTS;      // 16*64
    float* rowsq = Xr + TILE * DIM;         // 16
    int*   nn    = (int*)(rowsq + TILE);    // 16*20

    const int tid  = threadIdx.x;
    const int lane = tid & 31;
    const int wv   = tid >> 5;      // 8 waves
    const int half = lane >> 4;     // 0 or 1
    const int nm   = lane & 15;     // row/col within 16-tile

    const int b  = blockIdx.x >> 8;     // 256 row-tiles per batch
    const int rt = blockIdx.x & 255;
    const int i0 = rt * TILE;

    const size_t xrow_base = ((size_t)b * NPTS + i0) * DIM;

    // ---- Phase 0: stage 16-row block into LDS, compute row norms ----
    for (int e = tid; e < TILE * DIM; e += 256)
        Xr[e] = xg[xrow_base + e];
    __syncthreads();
    if (tid < TILE) {
        float s = 0.f;
        for (int c = 0; c < DIM; ++c) { float v = Xr[tid * DIM + c]; s += v * v; }
        rowsq[tid] = s;
    }
    __syncthreads();

    // ---- Build A fragments (f32 -> hi/lo f16 split) in contiguous k = 16*half + e order ----
    V16H aHi[2], aLo[2];
    for (int kb = 0; kb < 2; ++kb) {
        for (int e = 0; e < 16; ++e) {
            int p = e >> 1, s = e & 1;
            int kk = 16 * half + 2 * p + s;
            float v = Xr[nm * DIM + kb * 32 + kk];
            _Float16 h = (_Float16)v;
            aHi[kb].e[e] = h;
            aLo[kb].e[e] = (_Float16)(v - (float)h);
        }
    }
    float rsq[8];
    for (int v = 0; v < 8; ++v) rsq[v] = rowsq[v + 8 * half];

    // ---- Phase 1: sweep 256 column tiles; each wave owns every 8th tile ----
    for (int ct = wv; ct < NPTS / TILE; ct += 8) {
        const int j0 = ct * TILE;
        const float* xc = xg + ((size_t)b * NPTS + j0 + nm) * DIM;

        V16H bHi[2], bLo[2];
        float csq = 0.f;   // this lane covers half of K for column nm
        for (int kb = 0; kb < 2; ++kb) {
            for (int e = 0; e < 16; ++e) {
                int p = e >> 1, s = e & 1;
                int K = kb * 32 + 16 * half + 2 * p + s;   // B: contiguous k order, matches A
                float v = xc[K];
                csq += v * v;
                _Float16 h = (_Float16)v;
                bHi[kb].e[e] = h;
                bLo[kb].e[e] = (_Float16)(v - (float)h);
            }
        }
        csq += __shfl_xor(csq, 16, 32);   // pair lanes -> full column norm

        V8F acc = {};
        for (int kb = 0; kb < 2; ++kb) {
            acc.v = __builtin_amdgcn_wmma_f32_16x16x32_f16(false, aHi[kb].v, false, bHi[kb].v,
                                                           (short)0, acc.v, false, false);
            acc.v = __builtin_amdgcn_wmma_f32_16x16x32_f16(false, aHi[kb].v, false, bLo[kb].v,
                                                           (short)0, acc.v, false, false);
            acc.v = __builtin_amdgcn_wmma_f32_16x16x32_f16(false, aLo[kb].v, false, bHi[kb].v,
                                                           (short)0, acc.v, false, false);
        }

        // dist[i,j] = ||xi||^2 + ||xj||^2 - 2 <xi,xj>; C/D: M = v + 8*half, N = nm
        for (int v = 0; v < 8; ++v) {
            int M = v + 8 * half;
            dist[M * NPTS + j0 + nm] = rsq[v] + csq - 2.0f * acc.e[v];
        }
    }
    __syncthreads();

    // ---- Phase 2: per-row top-20 by iterative lexicographic-min extraction ----
    // Wave wv owns rows wv and wv+8 (wave-private -> no barriers needed inside).
    for (int rr = 0; rr < 2; ++rr) {
        const int r = wv + rr * 8;
        float* drow = dist + r * NPTS;
        for (int t = 0; t < KNN; ++t) {
            float bd = 3.4e38f; int bi = NPTS;
            for (int j = lane; j < NPTS; j += 32) {
                float d = drow[j];
                if (d < bd || (d == bd && j < bi)) { bd = d; bi = j; }
            }
            for (int off = 16; off > 0; off >>= 1) {
                float od = __shfl_xor(bd, off, 32);
                int   oi = __shfl_xor(bi, off, 32);
                if (od < bd || (od == bd && oi < bi)) { bd = od; bi = oi; }
            }
            if (lane == 0) { nn[r * KNN + t] = bi; drow[bi] = 3.4e38f; }
        }
    }
    __syncthreads();

    // ---- Phase 3: gather neighbors + write [16][20][128] output slab ----
    for (int p = wv; p < TILE * KNN; p += 8) {
        const int r = p / KNN, t = p % KNN;
        const int j = nn[r * KNN + t];
        const int c = lane * 2;                      // 32 lanes x 2 dims = 64
        const float t0 = Xr[r * DIM + c];
        const float t1 = Xr[r * DIM + c + 1];
        const float* xn = xg + ((size_t)b * NPTS + j) * DIM;
        const float n0 = xn[c], n1 = xn[c + 1];
        const size_t ob = (((size_t)b * NPTS + i0 + r) * KNN + t) * (size_t)(2 * DIM);
        out[ob + c]           = t0;
        out[ob + c + 1]       = t1;
        out[ob + DIM + c]     = n0 - t0;
        out[ob + DIM + c + 1] = n1 - t1;
    }
}

extern "C" void kernel_launch(void* const* d_in, const int* in_sizes, int n_in,
                              void* d_out, int out_size, void* d_ws, size_t ws_size,
                              hipStream_t stream) {
    (void)in_sizes; (void)n_in; (void)d_ws; (void)ws_size; (void)out_size;
    const float* x = (const float*)d_in[0];   // [8, 4096, 64] f32  (d_in[1] is k==20)
    float* out = (float*)d_out;               // [8, 4096, 20, 128] f32

    const size_t shmem = (size_t)(TILE * NPTS + TILE * DIM + TILE) * sizeof(float)
                       + (size_t)(TILE * KNN) * sizeof(int);   // ~267.6 KB < 320 KB WGP LDS
    dim3 grid(8 * (NPTS / TILE));   // 2048 blocks: (batch, row-tile)
    dim3 block(256);                // 8 wave32s
    edge_feature_knn_kernel<<<grid, block, shmem, stream>>>(x, out);
}
